// GAT_17635135717632
// MI455X (gfx1250) — matrix operand order repaired and checked by reading, not compiled, not run on hardware
//
#include <hip/hip_runtime.h>
#include <hip/hip_bf16.h>

typedef _Float16 v16h __attribute__((ext_vector_type(16)));
typedef _Float16 v8h  __attribute__((ext_vector_type(8)));
typedef float    v8f  __attribute__((ext_vector_type(8)));

// ---------- helpers ----------
__device__ __forceinline__ float lrelu(float x) { return x > 0.f ? x : 0.2f * x; }

// order-preserving float<->uint map for atomic max on floats (incl. negatives)
__device__ __forceinline__ unsigned f2o(float f) {
    unsigned u = __float_as_uint(f);
    return (u & 0x80000000u) ? ~u : (u | 0x80000000u);
}
__device__ __forceinline__ float o2f(unsigned k) {
    unsigned u = (k & 0x80000000u) ? (k ^ 0x80000000u) : ~k;
    return __uint_as_float(u);
}

// ---------- init / convert ----------
__global__ void fill_f32_k(float* p, float v, long long n) {
    long long t = (long long)blockIdx.x * blockDim.x + threadIdx.x;
    if (t < n) p[t] = v;
}
__global__ void fill_u32_k(unsigned* p, unsigned v, long long n) {
    long long t = (long long)blockIdx.x * blockDim.x + threadIdx.x;
    if (t < n) p[t] = v;
}
__global__ void cvt_f16_k(const float* __restrict__ in, _Float16* __restrict__ out, long long n) {
    long long t = (long long)blockIdx.x * blockDim.x + threadIdx.x;
    if (t < n) out[t] = (_Float16)in[t];
}

// Pre-swizzle W [Kout,Kin] (row major, f32) into the CDNA5 WMMA B-fragment order:
// Bp[((ot*(Kin/32)+kt)*32 + lane)*16 + e] = W[ot*16+(lane%16), kt*32 + 8*(lane/16) + (e%8) + 16*(e/8)]
// so each GEMM lane loads its v16h B fragment as one contiguous 32B read.
__global__ void pack_w_k(const float* __restrict__ W, _Float16* __restrict__ Bp,
                         int Kout, int Kin) {
    int t = blockIdx.x * blockDim.x + threadIdx.x;
    int KT = Kin >> 5;                   // k-tiles of 32
    int total = (Kout >> 4) * KT * 32;   // (o-tiles) * (k-tiles) * lanes
    if (t >= total) return;
    int lane = t & 31;
    int kt   = (t >> 5) % KT;
    int ot   = (t >> 5) / KT;
    int kcol = kt * 32 + 8 * (lane >> 4);
    _Float16* dst = Bp + (size_t)t * 16;
#pragma unroll
    for (int e = 0; e < 16; ++e)
        dst[e] = (_Float16)W[(size_t)(ot * 16 + (lane & 15)) * Kin + kcol + (e & 7) + 16 * (e >> 3)];
}

// ---------- WMMA GEMM: Hout[N,Kout] = A[N,Kin] (f16) @ W^T (pre-swizzled f16) ----------
// One wave computes a 16x64 output tile: 4 accumulators share one A fragment,
// giving 4 back-to-back v_wmma per K-step and 4x A reuse.
__global__ void gemm_wmma_k(const _Float16* __restrict__ A, const _Float16* __restrict__ Bp,
                            float* __restrict__ Hout, int Nrows, int Kin, int Kout) {
    int lane = threadIdx.x & 31;
    int gw   = (blockIdx.x * blockDim.x + threadIdx.x) >> 5;
    int numOG = Kout >> 6;               // groups of 4 o-tiles (64 cols)
    int numNT = Nrows >> 4;
    if (gw >= numOG * numNT) return;
    int og = gw % numOG;
    int nt = gw / numOG;

    int g = lane >> 4;        // half-wave group
    int m = lane & 15;        // row within tile (A) / column (C/D)
    int KT = Kin >> 5;

    const _Float16* arow = A + (size_t)(nt * 16 + m) * Kin + 8 * g;
    const _Float16* bb0  = Bp + (((size_t)(og * 4 + 0) * KT) * 32 + lane) * 16;
    const _Float16* bb1  = Bp + (((size_t)(og * 4 + 1) * KT) * 32 + lane) * 16;
    const _Float16* bb2  = Bp + (((size_t)(og * 4 + 2) * KT) * 32 + lane) * 16;
    const _Float16* bb3  = Bp + (((size_t)(og * 4 + 3) * KT) * 32 + lane) * 16;

    v8f c0 = {}, c1 = {}, c2 = {}, c3 = {};
    for (int kt = 0; kt < KT; ++kt) {
        // A fragment: e=0..7 -> k = kt*32 + 8g + e ; e=8..15 -> k = kt*32 + 16 + 8g + (e-8)
        v8h a0 = *(const v8h*)(arow + (size_t)kt * 32);
        v8h a1 = *(const v8h*)(arow + (size_t)kt * 32 + 16);
        v16h a;
#pragma unroll
        for (int e = 0; e < 8; ++e) { a[e] = a0[e]; a[8 + e] = a1[e]; }
        v16h b0 = *(const v16h*)(bb0 + (size_t)kt * 512);
        v16h b1 = *(const v16h*)(bb1 + (size_t)kt * 512);
        v16h b2 = *(const v16h*)(bb2 + (size_t)kt * 512);
        v16h b3 = *(const v16h*)(bb3 + (size_t)kt * 512);
        // speculative near-scope prefetch of next K-step (silently dropped past end)
        __builtin_prefetch(arow + (size_t)(kt + 1) * 32, 0, 3);
        __builtin_prefetch(bb0 + (size_t)(kt + 1) * 512, 0, 3);
        __builtin_prefetch(bb1 + (size_t)(kt + 1) * 512, 0, 3);
        __builtin_prefetch(bb2 + (size_t)(kt + 1) * 512, 0, 3);
        __builtin_prefetch(bb3 + (size_t)(kt + 1) * 512, 0, 3);
        c0 = __builtin_amdgcn_wmma_f32_16x16x32_f16(false, a, false, b0, (short)0, c0, false, false);
        c1 = __builtin_amdgcn_wmma_f32_16x16x32_f16(false, a, false, b1, (short)0, c1, false, false);
        c2 = __builtin_amdgcn_wmma_f32_16x16x32_f16(false, a, false, b2, (short)0, c2, false, false);
        c3 = __builtin_amdgcn_wmma_f32_16x16x32_f16(false, a, false, b3, (short)0, c3, false, false);
    }
    // C/D layout: VGPR r -> M = r + 8*(lane/16), N = lane%16
    float* out = Hout + (size_t)(nt * 16 + 8 * g) * Kout + og * 64 + m;
#pragma unroll
    for (int r = 0; r < 8; ++r) {
        out[(size_t)r * Kout +  0] = c0[r];
        out[(size_t)r * Kout + 16] = c1[r];
        out[(size_t)r * Kout + 32] = c2[r];
        out[(size_t)r * Kout + 48] = c3[r];
    }
}

// ---------- per-node attention scores ----------
__global__ void scores_k(const float* __restrict__ h, const float* __restrict__ avS,
                         const float* __restrict__ avD, float* __restrict__ outS,
                         float* __restrict__ outD, int n, int H, int C) {
    int t = blockIdx.x * blockDim.x + threadIdx.x;
    if (t >= n * H) return;
    int node = t / H, hd = t % H;
    const float* hp = h + (size_t)node * H * C + (size_t)hd * C;
    const float* as = avS + hd * C;
    const float* ad = avD + hd * C;
    float s = 0.f, d = 0.f;
    for (int c = 0; c < C; ++c) { float v = hp[c]; s += v * as[c]; d += v * ad[c]; }
    outS[t] = s;
    outD[t] = d;
}

// ---------- edge pass 1: segmented max ----------
__global__ void edge_max_k(const int* __restrict__ ei, int E, int n,
                           const float* __restrict__ aS, const float* __restrict__ aD,
                           unsigned* __restrict__ mkey, int H) {
    long long t = (long long)blockIdx.x * blockDim.x + threadIdx.x;
    long long tot = (long long)(E + n) * H;
    if (t >= tot) return;
    int i = (int)(t / H), hd = (int)(t % H);
    int src = i < E ? ei[i]     : i - E;
    int dst = i < E ? ei[E + i] : i - E;
    float logit = lrelu(aS[src * H + hd] + aD[dst * H + hd]);
    atomicMax(&mkey[dst * H + hd], f2o(logit));
}

// ---------- edge pass 2: segmented sum of exp ----------
__global__ void edge_sum_k(const int* __restrict__ ei, int E, int n,
                           const float* __restrict__ aS, const float* __restrict__ aD,
                           const unsigned* __restrict__ mkey, float* __restrict__ ssum, int H) {
    long long t = (long long)blockIdx.x * blockDim.x + threadIdx.x;
    long long tot = (long long)(E + n) * H;
    if (t >= tot) return;
    int i = (int)(t / H), hd = (int)(t % H);
    int src = i < E ? ei[i]     : i - E;
    int dst = i < E ? ei[E + i] : i - E;
    float logit = lrelu(aS[src * H + hd] + aD[dst * H + hd]);
    float mm = o2f(mkey[dst * H + hd]);
    atomicAdd(&ssum[dst * H + hd], expf(logit - mm));
}

// ---------- edge pass 3: weighted scatter (one wave per edge) ----------
__global__ void edge_scatter_k(const int* __restrict__ ei, int E, int n,
                               const float* __restrict__ aS, const float* __restrict__ aD,
                               const unsigned* __restrict__ mkey, const float* __restrict__ ssum,
                               const float* __restrict__ hfeat, float* __restrict__ agg,
                               int Kout, int H, int C) {
    int wid  = (blockIdx.x * blockDim.x + threadIdx.x) >> 5;
    int lane = threadIdx.x & 31;
    int ET = E + n;
    if (wid >= ET) return;
    int src = wid < E ? ei[wid]     : wid - E;
    int dst = wid < E ? ei[E + wid] : wid - E;
    float alpha[4];
    for (int hd = 0; hd < H; ++hd) {
        float logit = lrelu(aS[src * H + hd] + aD[dst * H + hd]);
        float mm = o2f(mkey[dst * H + hd]);
        float e = expf(logit - mm);
        alpha[hd] = e / (ssum[dst * H + hd] + 1e-16f);
    }
    const float* hs = hfeat + (size_t)src * Kout;
    float*       od = agg   + (size_t)dst * Kout;
    for (int ch = lane; ch < Kout; ch += 32)
        atomicAdd(&od[ch], alpha[ch / C] * hs[ch]);
}

// ---------- bias + optional relu ----------
__global__ void finalize_k(const float* __restrict__ agg, const float* __restrict__ bias,
                           float* __restrict__ out, long long total, int Kout, int doRelu) {
    long long t = (long long)blockIdx.x * blockDim.x + threadIdx.x;
    if (t >= total) return;
    int j = (int)(t % Kout);
    float v = agg[t] + bias[j];
    out[t] = doRelu ? fmaxf(v, 0.f) : v;
}

// ---------- host driver ----------
static inline int blk(long long n, int b) { return (int)((n + b - 1) / b); }

extern "C" void kernel_launch(void* const* d_in, const int* in_sizes, int n_in,
                              void* d_out, int out_size, void* d_ws, size_t ws_size,
                              hipStream_t stream) {
    const float* x   = (const float*)d_in[0];
    const int*   ei  = (const int*)d_in[1];
    const float* W1  = (const float*)d_in[2];
    const float* as1 = (const float*)d_in[3];
    const float* ad1 = (const float*)d_in[4];
    const float* b1  = (const float*)d_in[5];
    const float* W2  = (const float*)d_in[6];
    const float* as2 = (const float*)d_in[7];
    const float* ad2 = (const float*)d_in[8];
    const float* b2  = (const float*)d_in[9];
    const float* W3  = (const float*)d_in[10];
    const float* as3 = (const float*)d_in[11];
    const float* ad3 = (const float*)d_in[12];
    const float* b3  = (const float*)d_in[13];

    const int IN = 128, HID = 64, HEADS = 4;
    const int N = in_sizes[0] / IN;           // 50000 (divisible by 16)
    const int E = in_sizes[1] / 2;            // 800000
    const int KMAX = HEADS * HID;             // 256

    // workspace carve-out (256B aligned slices)
    char* base = (char*)d_ws;
    size_t off = 0;
    auto carve = [&](size_t bytes) {
        size_t o = off;
        off = (off + bytes + 255) & ~(size_t)255;
        return (void*)(base + o);
    };
    float*    feat  = (float*)carve((size_t)N * KMAX * 4);    // layer I/O features
    float*    hfeat = (float*)carve((size_t)N * KMAX * 4);    // GEMM output
    float*    agg   = (float*)carve((size_t)N * KMAX * 4);    // scatter accumulator
    _Float16* a16   = (_Float16*)carve((size_t)N * KMAX * 2); // f16 activations
    _Float16* w16p  = (_Float16*)carve((size_t)KMAX * KMAX * 2); // packed f16 W
    float*    asb   = (float*)carve((size_t)N * HEADS * 4);
    float*    adb   = (float*)carve((size_t)N * HEADS * 4);
    unsigned* mkey  = (unsigned*)carve((size_t)N * HEADS * 4);
    float*    ssum  = (float*)carve((size_t)N * HEADS * 4);
    (void)ws_size; (void)n_in;

    auto run_layer = [&](const float* fin, int Kin, const float* W, const float* avS,
                         const float* avD, const float* bias, int Kout, int H,
                         int doRelu, float* fout) {
        int C = Kout / H;
        long long nin  = (long long)N * Kin;
        long long nout = (long long)N * Kout;
        long long nh   = (long long)N * H;
        long long etH  = (long long)(E + N) * H;
        long long etW  = (long long)(E + N) * 32;

        cvt_f16_k<<<blk(nin, 256), 256, 0, stream>>>(fin, a16, nin);
        {
            int total = (Kout >> 4) * (Kin >> 5) * 32;
            pack_w_k<<<blk(total, 256), 256, 0, stream>>>(W, w16p, Kout, Kin);
        }
        fill_f32_k<<<blk(nout, 256), 256, 0, stream>>>(agg, 0.f, nout);
        fill_u32_k<<<blk(nh, 256), 256, 0, stream>>>(mkey, 0u, nh);
        fill_f32_k<<<blk(nh, 256), 256, 0, stream>>>(ssum, 0.f, nh);

        {
            long long waves = (long long)(N >> 4) * (Kout >> 6);
            gemm_wmma_k<<<blk(waves * 32, 256), 256, 0, stream>>>(a16, w16p, hfeat,
                                                                  N, Kin, Kout);
        }
        scores_k<<<blk(nh, 256), 256, 0, stream>>>(hfeat, avS, avD, asb, adb, N, H, C);
        edge_max_k<<<blk(etH, 256), 256, 0, stream>>>(ei, E, N, asb, adb, mkey, H);
        edge_sum_k<<<blk(etH, 256), 256, 0, stream>>>(ei, E, N, asb, adb, mkey, ssum, H);
        edge_scatter_k<<<blk(etW, 256), 256, 0, stream>>>(ei, E, N, asb, adb, mkey, ssum,
                                                          hfeat, agg, Kout, H, C);
        finalize_k<<<blk(nout, 256), 256, 0, stream>>>(agg, bias, fout, nout, Kout, doRelu);
    };

    // layer 1: 128 -> 4x64 concat, relu
    run_layer(x, IN, W1, as1, ad1, b1, HEADS * HID, HEADS, 1, feat);
    // layer 2: 256 -> 4x64 concat, relu
    run_layer(feat, HEADS * HID, W2, as2, ad2, b2, HEADS * HID, HEADS, 1, feat);
    // layer 3: 256 -> 64, 1 head, no relu, straight to output
    run_layer(feat, HEADS * HID, W3, as3, ad3, b3, HID, 1, 0, (float*)d_out);
}
